// StackedRNNCells_60567628808783
// MI455X (gfx1250) — matrix operand order repaired and checked by reading, not compiled, not run on hardware
//
#include <hip/hip_runtime.h>
#include <hip/hip_bf16.h>
#include <math.h>

// Stacked tanh-RNN on MI455X (gfx1250).
// Wavefront-diagonal scheduling; split-bf16 (hi+lo) WMMA with f32 accumulation.
// Weights pre-converted to bf16 hi/lo once per call; recurrent state kept in
// bf16 hi/lo so GEMM cells load WMMA fragments straight from global memory
// (no LDS, no barriers in the hot loop).

#define NL   8      // layers
#define HD   1024   // hidden
#define BT   64     // batch
#define TT   256    // seq len
#define KC   32     // WMMA K depth (bf16)

#define WPLANE  ((size_t)NL * HD * HD)   // 8,388,608 elems per weight plane
#define HPLANE  ((size_t)NL * BT * HD)   // 524,288 elems per state plane
#define FINOFF  ((size_t)BT * TT * HD)   // offset of h_final inside d_out

typedef __attribute__((ext_vector_type(16))) __bf16    v16bf;
typedef __attribute__((ext_vector_type(8)))  float     v8f;
typedef __attribute__((ext_vector_type(4)))  unsigned  u32x4;

struct Frag { union { unsigned short s[16]; u32x4 u[2]; v16bf v; }; };

__device__ __forceinline__ unsigned short bf16_rne(float f) {
    unsigned u = __float_as_uint(f);
    u += 0x7FFFu + ((u >> 16) & 1u);      // round-to-nearest-even
    return (unsigned short)(u >> 16);
}
__device__ __forceinline__ float bf16_f32(unsigned short h) {
    return __uint_as_float(((unsigned)h) << 16);
}

// ---- fragment gathers (ISA 16-bit layouts, wave32) -------------------------
// A 16x32: lane(h=lane/16, r=lane%16) row M=r, K in [8h,8h+8) and [16+8h,16+8h+8)
__device__ __forceinline__ void ldA(Frag& f, const unsigned short* rowk, int h) {
    f.u[0] = *(const u32x4*)(rowk + 8 * h);
    f.u[1] = *(const u32x4*)(rowk + 16 + 8 * h);
}
// B 32x16: lane holds col N=r, K in [8h,8h+8) and [16+8h,16+8h+8)
__device__ __forceinline__ void ldB(Frag& f, const unsigned short* rowk, int h) {
    f.u[0] = *(const u32x4*)(rowk + 8 * h);
    f.u[1] = *(const u32x4*)(rowk + 16 + 8 * h);
}
// A from f32 source (layer-0 input x), converted in-register to hi/lo bf16.
__device__ __forceinline__ void ldA_cvt(Frag& hi, Frag& lo, const float* rowk, int h) {
    float4 f0 = *(const float4*)(rowk + 8 * h);
    float4 f1 = *(const float4*)(rowk + 8 * h + 4);
    float4 f2 = *(const float4*)(rowk + 16 + 8 * h);
    float4 f3 = *(const float4*)(rowk + 16 + 8 * h + 4);
    float e[16] = { f0.x, f0.y, f0.z, f0.w, f1.x, f1.y, f1.z, f1.w,
                    f2.x, f2.y, f2.z, f2.w, f3.x, f3.y, f3.z, f3.w };
    #pragma unroll
    for (int i = 0; i < 16; ++i) {
        unsigned short hh = bf16_rne(e[i]);
        hi.s[i] = hh;
        lo.s[i] = bf16_rne(e[i] - bf16_f32(hh));
    }
}

// ---- one GEMM phase: acc[j] += Inp(16x1024) @ W[n-tile j]^T ----------------
// a_hi/a_lo: bf16 planes at this lane's A row (+k indexing), or af32 if CVT.
// w_hi/w_lo: weight planes at row (l*HD + n0 + r), j-th tile at +16*j*HD.
template <bool CVT>
__device__ __forceinline__ void run_phase(v8f* acc,
        const float* __restrict__ af32,
        const unsigned short* __restrict__ a_hi,
        const unsigned short* __restrict__ a_lo,
        const unsigned short* __restrict__ w_hi,
        const unsigned short* __restrict__ w_lo,
        int half)
{
    for (int k0 = 0; k0 < HD; k0 += KC) {
        Frag ah, al;
        if (CVT) {
            ldA_cvt(ah, al, af32 + k0, half);
        } else {
            ldA(ah, a_hi + k0, half);
            ldA(al, a_lo + k0, half);
        }
        #pragma unroll
        for (int j = 0; j < 4; ++j) {
            Frag bh, bl;
            const size_t off = (size_t)(16 * j) * HD + k0;
            ldB(bh, w_hi + off, half);
            ldB(bl, w_lo + off, half);
            // hi*hi + hi*lo + lo*hi (lo*lo ~2^-16 relative, dropped)
            acc[j] = __builtin_amdgcn_wmma_f32_16x16x32_bf16(false, ah.v, false, bh.v,
                                                             (short)0, acc[j], false, false);
            acc[j] = __builtin_amdgcn_wmma_f32_16x16x32_bf16(false, ah.v, false, bl.v,
                                                             (short)0, acc[j], false, false);
            acc[j] = __builtin_amdgcn_wmma_f32_16x16x32_bf16(false, al.v, false, bh.v,
                                                             (short)0, acc[j], false, false);
        }
    }
}

// ---- helper kernels --------------------------------------------------------
__global__ void __launch_bounds__(256)
cvt_weights_kernel(const float* __restrict__ src,
                   unsigned short* __restrict__ hi,
                   unsigned short* __restrict__ lo) {
    size_t i = ((size_t)blockIdx.x * 256 + threadIdx.x) * 4;
    float4 v = *(const float4*)(src + i);
    unsigned short h0 = bf16_rne(v.x), h1 = bf16_rne(v.y);
    unsigned short h2 = bf16_rne(v.z), h3 = bf16_rne(v.w);
    unsigned short l0 = bf16_rne(v.x - bf16_f32(h0));
    unsigned short l1 = bf16_rne(v.y - bf16_f32(h1));
    unsigned short l2 = bf16_rne(v.z - bf16_f32(h2));
    unsigned short l3 = bf16_rne(v.w - bf16_f32(h3));
    uint2 hv = make_uint2((unsigned)h0 | ((unsigned)h1 << 16),
                          (unsigned)h2 | ((unsigned)h3 << 16));
    uint2 lv = make_uint2((unsigned)l0 | ((unsigned)l1 << 16),
                          (unsigned)l2 | ((unsigned)l3 << 16));
    *(uint2*)(hi + i) = hv;
    *(uint2*)(lo + i) = lv;
}

__global__ void __launch_bounds__(256)
zero_u32_kernel(unsigned* p, int n) {
    int i = blockIdx.x * 256 + threadIdx.x;
    if (i < n) p[i] = 0u;
}

// ---- cell kernel: one wavefront diagonal -----------------------------------
// blockIdx.y -> cell (l = lmin + y, t = diag - l); blockIdx.x -> 64-wide N tile.
// Block = 128 threads (4 waves); wave w computes M rows [16w,16w+16) x N 64.
__global__ void __launch_bounds__(128)
rnn_cell_wmma_kernel(const float* __restrict__ x,
                     const unsigned short* __restrict__ wih_hi,
                     const unsigned short* __restrict__ wih_lo,
                     const unsigned short* __restrict__ whh_hi,
                     const unsigned short* __restrict__ whh_lo,
                     const float* __restrict__ bias_ih,
                     const float* __restrict__ bias_hh,
                     const unsigned short* __restrict__ hsp_hi,   // prev-parity hi plane
                     const unsigned short* __restrict__ hsp_lo,   // prev-parity lo plane
                     unsigned short* __restrict__ hsc_hi,         // cur-parity hi plane
                     unsigned short* __restrict__ hsc_lo,         // cur-parity lo plane
                     float* __restrict__ out,                     // (B,T,H) ++ (L,B,H)
                     int diag, int lmin)
{
    const int l    = lmin + blockIdx.y;
    const int t    = diag - l;
    const int n0   = blockIdx.x * 64;
    const int tid  = threadIdx.x;
    const int wave = tid >> 5;
    const int lane = tid & 31;
    const int half = lane >> 4;
    const int r    = lane & 15;
    const int m0   = wave * 16;

    v8f acc[4] = {{}, {}, {}, {}};

    // weight row bases for this lane: row (l*HD + n0 + r), tiles at +16*j*HD
    const size_t wrow = ((size_t)l * HD + n0 + r) * HD;
    const size_t arow = (size_t)(m0 + r);

    // Phase 0: layer input @ W_ih^T
    if (l == 0) {
        const float* af = x + (arow * TT + t) * HD;
        run_phase<true>(acc, af, nullptr, nullptr, wih_hi + wrow, wih_lo + wrow, half);
    } else {
        const size_t aoff = ((size_t)(l - 1) * BT + arow) * HD;
        run_phase<false>(acc, nullptr, hsp_hi + aoff, hsp_lo + aoff,
                         wih_hi + wrow, wih_lo + wrow, half);
    }
    // Phase 1: h_{t-1}^l @ W_hh^T
    {
        const size_t aoff = ((size_t)l * BT + arow) * HD;
        run_phase<false>(acc, nullptr, hsp_hi + aoff, hsp_lo + aoff,
                         whh_hi + wrow, whh_lo + wrow, half);
    }

    // Epilogue: bias + tanh; emit bf16 hi/lo state + f32 outputs.
    #pragma unroll
    for (int j = 0; j < 4; ++j) {
        const int o = n0 + 16 * j + r;
        const float bsum = bias_ih[l * HD + o] + bias_hh[l * HD + o];
        #pragma unroll
        for (int rr = 0; rr < 8; ++rr) {
            const int b = m0 + rr + 8 * half;          // C/D layout: M = reg + 8*(lane/16)
            const float val = tanhf(acc[j][rr] + bsum);
            const size_t sidx = ((size_t)l * BT + b) * HD + o;
            const unsigned short vh = bf16_rne(val);
            hsc_hi[sidx] = vh;
            hsc_lo[sidx] = bf16_rne(val - bf16_f32(vh));
            if (l == NL - 1) out[((size_t)b * TT + t) * HD + o] = val;  // outputs (B,T,H)
            if (t == TT - 1) out[FINOFF + sidx] = val;                  // h_final (L,B,H)
        }
    }
}

// ---- host orchestration ----------------------------------------------------
extern "C" void kernel_launch(void* const* d_in, const int* in_sizes, int n_in,
                              void* d_out, int out_size, void* d_ws, size_t ws_size,
                              hipStream_t stream) {
    const float* x       = (const float*)d_in[0];  // (B,T,H)
    const float* W_ih    = (const float*)d_in[1];  // (L,H,H)
    const float* W_hh    = (const float*)d_in[2];  // (L,H,H)
    const float* bias_ih = (const float*)d_in[3];  // (L,H)
    const float* bias_hh = (const float*)d_in[4];  // (L,H)
    float* out = (float*)d_out;

    // Workspace layout (ushorts): 4 weight planes (hi/lo x ih/hh) + 2 parity
    // state buffers, each {hi,lo} planes of (L,B,H). Total ~68 MB.
    unsigned short* wsu    = (unsigned short*)d_ws;
    unsigned short* wih_hi = wsu;
    unsigned short* wih_lo = wsu + WPLANE;
    unsigned short* whh_hi = wsu + 2 * WPLANE;
    unsigned short* whh_lo = wsu + 3 * WPLANE;
    unsigned short* hstate = wsu + 4 * WPLANE;     // 2 parities x 2 planes x HPLANE

    // 1) Pre-convert weights to bf16 hi/lo (done once, reused by all 2048 cells).
    {
        const int nblk = (int)(WPLANE / (256 * 4));  // exact: 8192
        cvt_weights_kernel<<<nblk, 256, 0, stream>>>(W_ih, wih_hi, wih_lo);
        cvt_weights_kernel<<<nblk, 256, 0, stream>>>(W_hh, whh_hi, whh_lo);
    }
    // 2) Zero the state buffers (bf16 zero == 0x0000).
    {
        const int nz = (int)(4 * HPLANE / 2);        // ushorts -> uints: 1,048,576
        zero_u32_kernel<<<nz / 256, 256, 0, stream>>>((unsigned*)hstate, nz);
    }
    // 3) Wavefront over diagonals d = t + l.
    for (int d = 0; d < TT + NL - 1; ++d) {
        const int lmin  = (d > TT - 1) ? (d - (TT - 1)) : 0;
        const int lmax  = (d < NL - 1) ? d : (NL - 1);
        const int cells = lmax - lmin + 1;
        unsigned short* hprev = hstate + (size_t)((d + 1) & 1) * 2 * HPLANE;
        unsigned short* hcur  = hstate + (size_t)(d & 1) * 2 * HPLANE;
        dim3 grid(HD / 64, cells);
        rnn_cell_wmma_kernel<<<grid, 128, 0, stream>>>(
            x, wih_hi, wih_lo, whh_hi, whh_lo, bias_ih, bias_hh,
            hprev, hprev + HPLANE,     // prev hi, lo
            hcur,  hcur + HPLANE,      // cur  hi, lo
            out, d, lmin);
    }
}
